// MemoryMoE_73967926772422
// MI455X (gfx1250) — matrix operand order repaired and checked by reading, not compiled, not run on hardware
//
#include <hip/hip_runtime.h>
#include <math.h>

// Problem constants (from reference): B=4, S=8192, D=512, E=8, DH=1024
#define BB   4
#define SS   8192
#define DD   512
#define DHH  1024
#define EE   8
#define TT   (BB * SS)                 // 32768 tokens
#define TILES_PER_E ((TT + 15) / 16)   // 2048 16-token tiles per expert (worst case)

typedef __bf16 bf16_t;
typedef __attribute__((ext_vector_type(16))) __bf16 v16bf;
typedef __attribute__((ext_vector_type(8)))  __bf16 v8bf;
typedef __attribute__((ext_vector_type(4)))  __bf16 v4bf;
typedef __attribute__((ext_vector_type(8)))  float  v8f;

// Branchless exact-GeLU: 0.5*v*(1+erf(v/sqrt(2))) with Abramowitz–Stegun
// 7.1.26 erf (|abs err| < 1.5e-7 — far below the bf16 input quantization).
// ~12 VALU ops, one v_rcp_f32 + one fast exp, no exec divergence.
__device__ __forceinline__ float gelu_exact(float v) {
  const float z = fabsf(v) * 0.70710678118654752f;     // |v|/sqrt(2)
  const float t = __builtin_amdgcn_rcpf(fmaf(0.3275911f, z, 1.0f));
  float p = fmaf(t, 1.061405429f, -1.453152027f);
  p = fmaf(t, p, 1.421413741f);
  p = fmaf(t, p, -0.284496736f);
  p = fmaf(t, p, 0.254829592f);
  const float erf_abs = fmaf(-(t * p), __expf(-z * z), 1.0f);  // in [0,1]
  const float erf_s   = copysignf(erf_abs, v);
  return 0.5f * v * (1.0f + erf_s);
}

// ---------------------------------------------------------------------------
// Pass 0: zero per-expert counters (workspace is not re-poisoned between calls)
// ---------------------------------------------------------------------------
__global__ void zero_counts(int* __restrict__ counts) {
  if (threadIdx.x < EE) counts[threadIdx.x] = 0;
}

// ---------------------------------------------------------------------------
// Pass 1: bucket token ids by routed expert (top-1).
// Bucket order is nondeterministic (atomics) but the final output is not:
// each token's full K-reduction and gating is independent of its tile slot.
// ---------------------------------------------------------------------------
__global__ void route_tokens(const int* __restrict__ idx,
                             int* __restrict__ counts,
                             int* __restrict__ bucket) {
  int t = blockIdx.x * blockDim.x + threadIdx.x;
  if (t >= TT) return;
  int e = idx[t];
  int r = atomicAdd(&counts[e], 1);
  bucket[e * TT + r] = t;
}

// ---------------------------------------------------------------------------
// Pass 2: repack fp32 row-major (E, K, N) weights into bf16 WMMA B-fragment
// order. Contiguous-K B-operand layout used here (not AMD's table order), 32x16 KxN:
//   lane l -> column n = (l & 15); K group = 16 * (l >> 4)
//   vector element j -> K = kt*32 + 16*(l>>4) + j   (16 contiguous K per lane)
// Flat packed index: (((e*KT + kt)*NT + nt)*32 + lane)*16 + j
// ---------------------------------------------------------------------------
template<int KT, int NT>
__global__ void pack_weights(const float* __restrict__ w, bf16_t* __restrict__ pw) {
  size_t i = (size_t)blockIdx.x * blockDim.x + threadIdx.x;
  const size_t total = (size_t)EE * KT * NT * 512;   // == E * K * N
  if (i >= total) return;
  int j    = (int)(i & 15);
  size_t r = i >> 4;
  int lane = (int)(r & 31); r >>= 5;
  int nt   = (int)(r % NT); r /= NT;
  int kt   = (int)(r % KT);
  size_t e = r / KT;
  int k = kt * 32 + ((lane >> 4) << 4) + j;
  int n = nt * 16 + (lane & 15);
  pw[i] = (bf16_t)w[(e * (size_t)(KT * 32) + (size_t)k) * (size_t)(NT * 16) + (size_t)n];
}

// ---------------------------------------------------------------------------
// Pass 3: per-expert 16-token-tile MLP:  y = gate * (gelu(X @ W0e) @ W1e)
// One 256-thread block (8 wave32) per tile. X and H staged in LDS as bf16.
// Each wave owns TWO 16-wide N tiles per K sweep: A fragment loaded once,
// two independent WMMA accumulator chains in flight.
// ---------------------------------------------------------------------------
__global__ __launch_bounds__(256)
void moe_expert_mlp(const float* __restrict__ x,
                    const float* __restrict__ rw,
                    const int*   __restrict__ counts,
                    const int*   __restrict__ bucket,
                    const bf16_t* __restrict__ pw0,
                    const bf16_t* __restrict__ pw1,
                    float* __restrict__ out) {
  const int e    = blockIdx.x / TILES_PER_E;
  const int tile = blockIdx.x % TILES_PER_E;
  const int cnt  = counts[e];
  if (tile * 16 >= cnt) return;   // uniform per block: safe w.r.t. barriers

  __shared__ __align__(16) bf16_t sX[16][DD];    // 16 KB
  __shared__ __align__(16) bf16_t sH[16][DHH];   // 32 KB
  __shared__ int   sTok[16];
  __shared__ float sGate[16];

  const int tid  = threadIdx.x;
  const int lane = tid & 31;
  const int wave = tid >> 5;
  const int m    = lane & 15;   // A row / D column-lane
  const int half = lane >> 4;

  if (tid < 16) {
    int rank = tile * 16 + tid;
    int t = -1; float g = 0.0f;
    if (rank < cnt) {
      t = bucket[e * TT + rank];
      g = rw[(size_t)t * EE + e];
    }
    sTok[tid]  = t;
    sGate[tid] = g;
  }
  __syncthreads();

  // Stage 16 gathered token rows: fp32 global (float4) -> bf16 LDS
  for (int i = tid; i < 16 * (DD / 4); i += 256) {
    int rrow = i >> 7;                 // DD/4 == 128 float4 per row
    int c4   = (i & 127) << 2;
    int t = sTok[rrow];
    float4 v = make_float4(0.f, 0.f, 0.f, 0.f);
    if (t >= 0) v = ((const float4*)(x + (size_t)t * DD))[c4 >> 2];
    v4bf p;
    p[0] = (bf16_t)v.x; p[1] = (bf16_t)v.y; p[2] = (bf16_t)v.z; p[3] = (bf16_t)v.w;
    *(v4bf*)&sX[rrow][c4] = p;
  }
  __syncthreads();

  // ---------------- GEMM1: H = gelu(X @ W0[e]),  M=16 N=1024 K=512 ----------
  // A fragment (16x32 bf16), contiguous K to match packed B: lane<16 holds K{0..15},
  // lane>=16 holds K{16..31} -> two contiguous 16B LDS loads.
  {
    const size_t ktStride = (size_t)(DHH / 16) * 32;  // v16bf per kt step
    const size_t ntStride = (size_t)8 * 32;           // v16bf between nt and nt+8
    for (int p = 0; p < 4; ++p) {                     // 8 tiles/wave -> 4 pairs
      const int nt0 = wave + 16 * p;                  // second tile: nt0 + 8
      v8f acc0 = {}, acc1 = {};
      const v16bf* bp = (const v16bf*)(pw0 +
          ((((size_t)e * (DD / 32)) * (DHH / 16) + (size_t)nt0) * 32 + (size_t)lane) * 16);
      #pragma unroll 4
      for (int kt = 0; kt < DD / 32; ++kt) {
        v8bf alo = *(const v8bf*)&sX[m][kt * 32 + half * 16];
        v8bf ahi = *(const v8bf*)&sX[m][kt * 32 + half * 16 + 8];
        v16bf a = __builtin_shufflevector(alo, ahi,
                    0,1,2,3,4,5,6,7,8,9,10,11,12,13,14,15);
        v16bf b0 = bp[(size_t)kt * ktStride];
        v16bf b1 = bp[(size_t)kt * ktStride + ntStride];
        __builtin_prefetch(bp + (size_t)(kt + 1) * ktStride, 0, 3); // global_prefetch_b8
        acc0 = __builtin_amdgcn_wmma_f32_16x16x32_bf16(
                   false, a, false, b0, (short)0, acc0, false, false);
        acc1 = __builtin_amdgcn_wmma_f32_16x16x32_bf16(
                   false, a, false, b1, (short)0, acc1, false, false);
      }
      // C/D layout: lane -> N = lane&15, vector elem r -> M = r + 8*(lane>>4)
      const int n0 = nt0 * 16 + m;
      #pragma unroll
      for (int r = 0; r < 8; ++r) {
        sH[r + 8 * half][n0]       = (bf16_t)gelu_exact(acc0[r]);
        sH[r + 8 * half][n0 + 128] = (bf16_t)gelu_exact(acc1[r]);
      }
    }
  }
  __syncthreads();

  // ---------------- GEMM2: OUT = H @ W1[e],  M=16 N=512 K=1024 --------------
  {
    const size_t ktStride = (size_t)(DD / 16) * 32;   // v16bf per kt step
    const size_t ntStride = (size_t)8 * 32;
    for (int p = 0; p < 2; ++p) {                     // 4 tiles/wave -> 2 pairs
      const int nt0 = wave + 16 * p;
      v8f acc0 = {}, acc1 = {};
      const v16bf* bp = (const v16bf*)(pw1 +
          ((((size_t)e * (DHH / 32)) * (DD / 16) + (size_t)nt0) * 32 + (size_t)lane) * 16);
      #pragma unroll 4
      for (int kt = 0; kt < DHH / 32; ++kt) {
        v8bf alo = *(const v8bf*)&sH[m][kt * 32 + half * 16];
        v8bf ahi = *(const v8bf*)&sH[m][kt * 32 + half * 16 + 8];
        v16bf a = __builtin_shufflevector(alo, ahi,
                    0,1,2,3,4,5,6,7,8,9,10,11,12,13,14,15);
        v16bf b0 = bp[(size_t)kt * ktStride];
        v16bf b1 = bp[(size_t)kt * ktStride + ntStride];
        __builtin_prefetch(bp + (size_t)(kt + 1) * ktStride, 0, 3);
        acc0 = __builtin_amdgcn_wmma_f32_16x16x32_bf16(
                   false, a, false, b0, (short)0, acc0, false, false);
        acc1 = __builtin_amdgcn_wmma_f32_16x16x32_bf16(
                   false, a, false, b1, (short)0, acc1, false, false);
      }
      const int n0 = nt0 * 16 + m;
      #pragma unroll
      for (int r = 0; r < 8; ++r) {
        int row = r + 8 * half;
        int t = sTok[row];
        if (t >= 0) {
          float g = sGate[row];
          out[(size_t)t * DD + n0]       = acc0[r] * g;
          out[(size_t)t * DD + n0 + 128] = acc1[r] * g;
        }
      }
    }
  }
}

// ---------------------------------------------------------------------------
// Launch. Workspace layout (needs ~17.8 MB):
//   [0,256)                       : per-expert counters (8 ints)
//   [256, 256+1MB)                : bucket, E*T token ids
//   [.., +8MB)                    : packed bf16 w0 fragments
//   [.., +8MB)                    : packed bf16 w1 fragments
// ---------------------------------------------------------------------------
extern "C" void kernel_launch(void* const* d_in, const int* in_sizes, int n_in,
                              void* d_out, int out_size, void* d_ws, size_t ws_size,
                              hipStream_t stream) {
  const float* x   = (const float*)d_in[0];   // (B,S,D) fp32
  const float* rw  = (const float*)d_in[1];   // (B,S,E) fp32
  const int*   idx = (const int*)  d_in[2];   // (B,S,1) int
  const float* w0  = (const float*)d_in[3];   // (E,D,DH) fp32
  const float* w1  = (const float*)d_in[4];   // (E,DH,D) fp32
  float* out = (float*)d_out;                 // (B,S,D) fp32

  char* ws = (char*)d_ws;
  int*    counts = (int*)ws;
  int*    bucket = (int*)(ws + 256);
  bf16_t* pw0 = (bf16_t*)(ws + 256 + (size_t)EE * TT * sizeof(int));
  bf16_t* pw1 = pw0 + (size_t)EE * DD * DHH;

  zero_counts<<<1, 32, 0, stream>>>(counts);
  route_tokens<<<(TT + 255) / 256, 256, 0, stream>>>(idx, counts, bucket);

  const size_t perW = (size_t)EE * DD * DHH;  // elements to pack per weight tensor
  pack_weights<DD / 32, DHH / 16><<<(unsigned)((perW + 255) / 256), 256, 0, stream>>>(w0, pw0);
  pack_weights<DHH / 32, DD / 16><<<(unsigned)((perW + 255) / 256), 256, 0, stream>>>(w1, pw1);

  moe_expert_mlp<<<EE * TILES_PER_E, 256, 0, stream>>>(
      x, rw, counts, bucket, pw0, pw1, out);
}
